// HammingSimilarities_65635690218322
// MI455X (gfx1250) — matrix operand order repaired and checked by reading, not compiled, not run on hardware
//
#include <hip/hip_runtime.h>

typedef __bf16 v4bf  __attribute__((ext_vector_type(4)));
typedef __bf16 v8bf  __attribute__((ext_vector_type(8)));
typedef __bf16 v16bf __attribute__((ext_vector_type(16)));
typedef float  v8f   __attribute__((ext_vector_type(8)));

#define BB 4
#define NN 1024
#define LL 512
#define AA 21
#define KK (LL * AA)   // 10752 = 168 * 64
#define GG 16
#define GS 64
#define KC 64          // K-chunk staged in LDS per iteration
#define SB 72          // LDS row stride (bf16 elems); 144B = 16B*odd -> bank-optimal b128 loads

__device__ __forceinline__ v16bf cat16(v8bf a, v8bf b) {
  return __builtin_shufflevector(a, b, 0, 1, 2, 3, 4, 5, 6, 7, 8, 9, 10, 11,
                                 12, 13, 14, 15);
}

// Gram partials via bf16 hi/lo split: S = Ahi*Bhi + Ahi*Blo + Alo*Bhi (fp32 acc).
// dst is either a 64x64 partial block (partial_mode=1, row_stride=64) in ws,
// or the final output block (partial_mode=0, row_stride=N) with scale applied.
__global__ __launch_bounds__(256) void gram_bf16x3_kernel(
    const float* __restrict__ x, float* __restrict__ dst,
    int ksplit, int row_stride, float scale, int partial_mode) {
  __shared__ __bf16 lhs_hi[GS * SB];
  __shared__ __bf16 lhs_lo[GS * SB];

  const int wg  = blockIdx.x;
  const int bg  = wg / ksplit;          // which (b,g) group
  const int spl = wg - bg * ksplit;     // which K-split
  const int b   = bg >> 4;
  const int g   = bg & 15;

  const int tid  = threadIdx.x;
  const int lane = tid & 31;
  const int wave = tid >> 5;

  // ---- global staging: thread loads 4 float4s of one row per chunk
  const int lrow = tid >> 2;            // 0..63 (row within group)
  const int lq   = tid & 3;
  const float* rowp = x + (size_t)(b * NN + g * GS + lrow) * KK;

  // ---- this wave's two output tiles (share the A row-block ti)
  const int t0  = 2 * wave;             // tile id 0..15 -> (ti, tj)
  const int ti  = t0 >> 2;
  const int tj0 = t0 & 3;
  const int tj1 = tj0 + 1;

  const int half = lane >> 4;
  const int r15  = lane & 15;
  // 16-bit A 16x32: lane (row M=r15) holds K = kk+16*half..+15, contiguous
  const int aoff = r15 * SB + 16 * half;   // + 16*t*SB + kk
  // 16-bit B 32x16: lane (col N=r15) holds K = kk+16*half..+15, contiguous
  const int boff = r15 * SB + 16 * half;   // + 16*t*SB + kk

  const int kchunks = (KK / KC) / ksplit;  // 168 / ksplit
  const int kbase   = spl * kchunks * KC;

  v8f c0 = {};
  v8f c1 = {};

  float4 stage[4];
#pragma unroll
  for (int v = 0; v < 4; ++v)
    stage[v] = *(const float4*)(rowp + kbase + (lq + 4 * v) * 4);

  for (int c = 0; c < kchunks; ++c) {
    const int k0 = kbase + c * KC;
    __syncthreads();  // previous chunk's compute done reading LDS
#pragma unroll
    for (int v = 0; v < 4; ++v) {
      const int col = (lq + 4 * v) * 4;
      float4 s = stage[v];
      // hi = bf16(x) (RTNE), lo = bf16(x - hi): drops only ~2^-18 rel. error
      __bf16 h0 = (__bf16)s.x, h1 = (__bf16)s.y, h2 = (__bf16)s.z,
             h3 = (__bf16)s.w;
      v4bf hv = {h0, h1, h2, h3};
      v4bf lv = {(__bf16)(s.x - (float)h0), (__bf16)(s.y - (float)h1),
                 (__bf16)(s.z - (float)h2), (__bf16)(s.w - (float)h3)};
      *(v4bf*)(&lhs_hi[lrow * SB + col]) = hv;
      *(v4bf*)(&lhs_lo[lrow * SB + col]) = lv;
    }
    __syncthreads();

    // overlap next global fetch with this chunk's WMMAs
    if (c + 1 < kchunks) {
#pragma unroll
      for (int v = 0; v < 4; ++v)
        stage[v] = *(const float4*)(rowp + k0 + KC + (lq + 4 * v) * 4);
    }

#pragma unroll
    for (int kk = 0; kk < KC; kk += 32) {
      const __bf16* pah = &lhs_hi[16 * ti * SB + aoff + kk];
      const __bf16* pal = &lhs_lo[16 * ti * SB + aoff + kk];
      v16bf ah = cat16(*(const v8bf*)(pah), *(const v8bf*)(pah + 8));
      v16bf al = cat16(*(const v8bf*)(pal), *(const v8bf*)(pal + 8));

      const __bf16* pbh0 = &lhs_hi[16 * tj0 * SB + boff + kk];
      const __bf16* pbl0 = &lhs_lo[16 * tj0 * SB + boff + kk];
      v16bf bh0 = cat16(*(const v8bf*)(pbh0), *(const v8bf*)(pbh0 + 8));
      v16bf bl0 = cat16(*(const v8bf*)(pbl0), *(const v8bf*)(pbl0 + 8));

      const __bf16* pbh1 = &lhs_hi[16 * tj1 * SB + boff + kk];
      const __bf16* pbl1 = &lhs_lo[16 * tj1 * SB + boff + kk];
      v16bf bh1 = cat16(*(const v8bf*)(pbh1), *(const v8bf*)(pbh1 + 8));
      v16bf bl1 = cat16(*(const v8bf*)(pbl1), *(const v8bf*)(pbl1 + 8));

      c0 = __builtin_amdgcn_wmma_f32_16x16x32_bf16(false, ah, false, bh0,
                                                   (short)0, c0, false, false);
      c1 = __builtin_amdgcn_wmma_f32_16x16x32_bf16(false, ah, false, bh1,
                                                   (short)0, c1, false, false);
      c0 = __builtin_amdgcn_wmma_f32_16x16x32_bf16(false, ah, false, bl0,
                                                   (short)0, c0, false, false);
      c1 = __builtin_amdgcn_wmma_f32_16x16x32_bf16(false, ah, false, bl1,
                                                   (short)0, c1, false, false);
      c0 = __builtin_amdgcn_wmma_f32_16x16x32_bf16(false, al, false, bh0,
                                                   (short)0, c0, false, false);
      c1 = __builtin_amdgcn_wmma_f32_16x16x32_bf16(false, al, false, bh1,
                                                   (short)0, c1, false, false);
    }
  }

  // ---- store: C/D layout M = vgpr + 8*(lane>=16), N = lane&15
  float* blk;
  if (partial_mode) {
    blk = dst + ((size_t)(bg * ksplit + spl) << 12);            // 64*64 block
  } else {
    blk = dst + (size_t)b * NN * NN + (size_t)(g * GS) * NN + g * GS;
  }
  const int ibase = ti * 16 + 8 * half;
#pragma unroll
  for (int v = 0; v < 8; ++v) {
    blk[(size_t)(ibase + v) * row_stride + tj0 * 16 + r15] = c0[v] * scale;
    blk[(size_t)(ibase + v) * row_stride + tj1 * 16 + r15] = c1[v] * scale;
  }
}

// Deterministic reduction of K-split partials into the diagonal blocks.
__global__ __launch_bounds__(256) void reduce_blocks_kernel(
    const float* __restrict__ ws, float* __restrict__ out, int ksplit) {
  const int t  = blockIdx.x * 256 + threadIdx.x;  // exactly 64*4096 threads
  const int bg = t >> 12;
  const int e  = t & 4095;
  float s = 0.0f;
  for (int k = 0; k < ksplit; ++k)
    s += ws[((size_t)(bg * ksplit + k) << 12) + e];
  const int b = bg >> 4, g = bg & 15;
  const int i = e >> 6, j = e & 63;
  out[(size_t)b * NN * NN + (size_t)(g * GS + i) * NN + (g * GS + j)] =
      s * (1.0f / (float)LL);
}

__global__ __launch_bounds__(256) void fill_nan_kernel(float4* __restrict__ out,
                                                       int n4) {
  const int i = blockIdx.x * 256 + threadIdx.x;
  const float nv = __builtin_nanf("");
  if (i < n4) out[i] = make_float4(nv, nv, nv, nv);
}

extern "C" void kernel_launch(void* const* d_in, const int* in_sizes, int n_in,
                              void* d_out, int out_size, void* d_ws,
                              size_t ws_size, hipStream_t stream) {
  (void)in_sizes; (void)n_in;
  const float* x = (const float*)d_in[0];
  float* out = (float*)d_out;

  // 1) NaN-fill the whole [B,N,N] output (blocks overwritten afterwards).
  const int n4 = out_size / 4;  // 1,048,576 float4
  fill_nan_kernel<<<(n4 + 255) / 256, 256, 0, stream>>>((float4*)out, n4);

  // 2) Gram matrices. Split K across workgroups to fill the chip, using
  //    d_ws for fp32 partials + deterministic reduce (no fp atomics).
  const size_t blk_bytes = (size_t)64 * 4096 * sizeof(float);  // per split-set
  int ksplit = 0;
  if (ws_size >= 8 * blk_bytes)      ksplit = 8;   // 512 WGs
  else if (ws_size >= 4 * blk_bytes) ksplit = 4;   // 256 WGs
  else if (ws_size >= 1 * blk_bytes) ksplit = 1;   // 64 WGs via ws
  if (ksplit > 0) {
    float* ws = (float*)d_ws;
    gram_bf16x3_kernel<<<64 * ksplit, 256, 0, stream>>>(x, ws, ksplit, 64,
                                                        1.0f, 1);
    reduce_blocks_kernel<<<(64 * 4096) / 256, 256, 0, stream>>>(ws, out,
                                                                ksplit);
  } else {
    // Fallback: single pass, write blocks directly with scale.
    gram_bf16x3_kernel<<<64, 256, 0, stream>>>(x, out, 1, NN,
                                               1.0f / (float)LL, 0);
  }
}
